// ResidualBlock_noBN_modulation_25786983645752
// MI455X (gfx1250) — matrix operand order repaired and checked by reading, not compiled, not run on hardware
//
#include <hip/hip_runtime.h>
#include <hip/hip_bf16.h>

typedef _Float16 half_t;
typedef __attribute__((ext_vector_type(16))) _Float16 v16h;
typedef __attribute__((ext_vector_type(8)))  float    v8f;

#define NF    64
#define EMB   512
#define KW    3
#define BATCH 16
#define HH    192
#define WW    192
#define SCALE (1.0f/24.0f)         /* 1/sqrt(64*9) */

#define NSTEP 18                    /* K = 64*9 = 576 = 18 steps of 32 */
#define ROWS  4                     /* output rows per block */
#define SROWS (ROWS + 2)            /* staged input rows */
#define PANEL_ELEMS (4*NSTEP*32*16) /* 36864 f16 per weight set */
#define PANEL_BYTES (PANEL_ELEMS*2)
#define TENSOR_HALF_BYTES ((size_t)BATCH*NF*HH*WW*2)

// Decode a flat panel element index into (outch, cin, ky, kx) following the
// contiguous-K 16x32 A-fragment lane layout used by this file (not the ISA table order):
//   lane L<16 : M=L,    K = {0..15}
//   lane L>=16: M=L-16, K = {16..31}
// K ordering within a step: cin = (s&1)*32 + K, tap = s>>1  (k = tap*64 + cin)
__device__ __forceinline__ void panel_decode(int e, int& o, int& cin, int& dy, int& dx) {
    int j    = e & 15;
    int lane = (e >> 4) & 31;
    int s    = (e >> 9) % NSTEP;
    int t    = e / (512 * NSTEP);
    int hlf  = lane >> 4;
    int m    = lane & 15;
    int kl   = hlf * 16 + j;
    cin = ((s & 1) << 5) + kl;
    int tap = s >> 1;
    dy = tap / 3;
    dx = tap % 3;
    o  = t * 16 + m;
}

__global__ void pack_static_weights(const float* __restrict__ w0,
                                    const float* __restrict__ w1,
                                    const float* __restrict__ w2,
                                    half_t* __restrict__ p0,
                                    half_t* __restrict__ p1,
                                    half_t* __restrict__ p2) {
    int e = blockIdx.x * blockDim.x + threadIdx.x;
    if (e >= PANEL_ELEMS) return;
    const float* w; half_t* p;
    if (blockIdx.y == 0)      { w = w0; p = p0; }
    else if (blockIdx.y == 1) { w = w1; p = p1; }
    else                      { w = w2; p = p2; }
    int o, cin, dy, dx;
    panel_decode(e, o, cin, dy, dx);
    p[e] = (half_t)w[((o * NF + cin) * KW + dy) * KW + dx];
}

__global__ void style_kernel(const float* __restrict__ emb,
                             const float* __restrict__ mod_w,
                             const float* __restrict__ mod_b,
                             float* __restrict__ style) {
    int t = blockIdx.x * blockDim.x + threadIdx.x;
    if (t >= BATCH * NF) return;
    int b = t / NF, c = t % NF;
    const float* e = emb + b * EMB;
    const float* w = mod_w + c * EMB;
    float s = mod_b[c];
    for (int k = 0; k < EMB; ++k) s += e[k] * w[k];
    style[t] = s;
}

__global__ void demod_kernel(const float* __restrict__ base_w,
                             const float* __restrict__ style,
                             float* __restrict__ demod) {
    int t = blockIdx.x * blockDim.x + threadIdx.x;
    if (t >= BATCH * NF) return;
    int b = t / NF, o = t % NF;
    float s = 0.f;
    for (int i = 0; i < NF; ++i) {
        float st = style[b * NF + i];
        for (int tap = 0; tap < 9; ++tap) {
            float wv = SCALE * base_w[(o * NF + i) * 9 + tap] * st;
            s += wv * wv;
        }
    }
    demod[t] = rsqrtf(s + 1e-8f);
}

__global__ void pack_mod_weights(const float* __restrict__ base_w,
                                 const float* __restrict__ style,
                                 const float* __restrict__ demod,
                                 half_t* __restrict__ pmod) {
    int e = blockIdx.x * blockDim.x + threadIdx.x;
    int b = blockIdx.y;
    if (e >= PANEL_ELEMS) return;
    int o, cin, dy, dx;
    panel_decode(e, o, cin, dy, dx);
    float wv = SCALE * base_w[((o * NF + cin) * KW + dy) * KW + dx]
             * style[b * NF + cin] * demod[b * NF + o];
    pmod[(size_t)b * PANEL_ELEMS + e] = (half_t)wv;
}

// Implicit-GEMM 3x3 conv, 4-row blocking.
// One block = 4 waves = all 64 outch for one (batch, 4-row, 16-pixel) tile.
// Each wave: 16 outch x 16 pixels x 4 rows; K=576 in 18 steps of 32; each
// panel load (A fragment) feeds 4 v_wmma_f32_16x16x32_f16 (one per row).
// act: 0=none, 1=relu, 2=leaky(0.1)
template <typename Tin, typename Tout>
__global__ __launch_bounds__(128)
void conv3x3_wmma(const Tin* __restrict__ in,
                  const half_t* __restrict__ panels,
                  int panel_batch_stride,
                  const float* __restrict__ bias,
                  const float* __restrict__ residual,   // f32 (original x) or null
                  const float* __restrict__ noise,
                  const float* __restrict__ wnoise,
                  int act,
                  Tout* __restrict__ out) {
    __shared__ __align__(32) half_t lds[SROWS * 18 * NF];

    const int w0  = blockIdx.x * 16;
    const int h0  = blockIdx.y * ROWS;
    const int b   = blockIdx.z;
    const int tid = threadIdx.x;

    // Stage the im2col patch: rows h0-1..h0+ROWS, cols w0-1..w0+16, all 64 cin,
    // as f16. LDS layout [row][px][cin] so a B fragment (16 contiguous cin at
    // one pixel) is one 32B ds load.
    const Tin* inb = in + (size_t)b * NF * HH * WW;
    for (int i = tid; i < SROWS * 18 * NF; i += 128) {
        int px  = i % 18;             // consecutive tids -> consecutive px: coalesced
        int cin = (i / 18) % NF;
        int dy  = i / (18 * NF);
        int gy  = h0 - 1 + dy;
        int gx  = w0 - 1 + px;
        float v = 0.f;
        if ((unsigned)gy < (unsigned)HH && (unsigned)gx < (unsigned)WW)
            v = (float)inb[(size_t)cin * HH * WW + (size_t)gy * WW + gx];
        lds[(dy * 18 + px) * NF + cin] = (half_t)v;
    }
    __syncthreads();

    const int wave = tid >> 5;
    const int lane = tid & 31;
    const int hlf  = lane >> 4;   // K-half selector for A/B fragments
    const int n    = lane & 15;   // output pixel within tile

    const half_t* pan = panels + (size_t)b * panel_batch_stride
                               + (size_t)wave * (NSTEP * 512);
    __builtin_prefetch(pan, 0, 1);   // global_prefetch_b8 on the weight panel

    v8f acc0 = {}, acc1 = {}, acc2 = {}, acc3 = {};
    #pragma unroll
    for (int s = 0; s < NSTEP; ++s) {
        const int tap  = s >> 1;
        const int dy   = tap / 3;
        const int dx   = tap % 3;
        const int cin0 = ((s & 1) << 5) + hlf * 16;  // B layout: lane holds K=hlf*16..+15
        const int pxo  = n + dx;

        const v16h a  = *(const v16h*)(pan + s * 512 + lane * 16);
        const v16h b0 = *(const v16h*)(&lds[((dy + 0) * 18 + pxo) * NF + cin0]);
        const v16h b1 = *(const v16h*)(&lds[((dy + 1) * 18 + pxo) * NF + cin0]);
        const v16h b2 = *(const v16h*)(&lds[((dy + 2) * 18 + pxo) * NF + cin0]);
        const v16h b3 = *(const v16h*)(&lds[((dy + 3) * 18 + pxo) * NF + cin0]);
        acc0 = __builtin_amdgcn_wmma_f32_16x16x32_f16(false, a, false, b0, (short)0, acc0, false, false);
        acc1 = __builtin_amdgcn_wmma_f32_16x16x32_f16(false, a, false, b1, (short)0, acc1, false, false);
        acc2 = __builtin_amdgcn_wmma_f32_16x16x32_f16(false, a, false, b2, (short)0, acc2, false, false);
        acc3 = __builtin_amdgcn_wmma_f32_16x16x32_f16(false, a, false, b3, (short)0, acc3, false, false);
    }

    // C/D layout: VGPR r -> M = hlf*8 + r, N = lane&15
    const int wcol  = w0 + n;
    const int mbase = hlf * 8;
    #pragma unroll
    for (int rr = 0; rr < ROWS; ++rr) {
        const v8f& acc = (rr == 0) ? acc0 : (rr == 1) ? acc1 : (rr == 2) ? acc2 : acc3;
        const int h = h0 + rr;
        float nz = 0.f;
        if (noise) nz = wnoise[0] * noise[((size_t)b * HH + h) * WW + wcol];
        #pragma unroll
        for (int r = 0; r < 8; ++r) {
            const int o = wave * 16 + mbase + r;
            float v = acc[r];
            if (bias) v += bias[o];
            const size_t idx = (((size_t)b * NF + o) * HH + h) * WW + wcol;
            if (residual) v += residual[idx];
            v += nz;
            if (act == 1)      v = v > 0.f ? v : 0.f;
            else if (act == 2) v = v >= 0.f ? v : 0.1f * v;
            out[idx] = (Tout)v;
        }
    }
}

extern "C" void kernel_launch(void* const* d_in, const int* in_sizes, int n_in,
                              void* d_out, int out_size, void* d_ws, size_t ws_size,
                              hipStream_t stream) {
    (void)in_sizes; (void)n_in; (void)out_size; (void)ws_size;

    const float* x       = (const float*)d_in[0];
    const float* embv    = (const float*)d_in[1];
    const float* noise   = (const float*)d_in[2];
    const float* conv1_w = (const float*)d_in[3];
    const float* conv1_b = (const float*)d_in[4];
    const float* conv2_w = (const float*)d_in[5];
    const float* conv2_b = (const float*)d_in[6];
    const float* mod_w   = (const float*)d_in[7];
    const float* mod_b   = (const float*)d_in[8];
    const float* base_w  = (const float*)d_in[9];   // [1,64,64,3,3]
    const float* wnoise  = (const float*)d_in[10];  // scalar
    const float* convl_w = (const float*)d_in[11];
    const float* convl_b = (const float*)d_in[12];
    float* out = (float*)d_out;

    char* ws = (char*)d_ws;
    float*  style = (float*)(ws);                                   // 4 KB
    float*  demod = (float*)(ws + 4096);                            // 4 KB
    half_t* pan1  = (half_t*)(ws + 8192);
    half_t* pan2  = (half_t*)(ws + 8192 + (size_t)PANEL_BYTES);
    half_t* panl  = (half_t*)(ws + 8192 + 2 * (size_t)PANEL_BYTES);
    half_t* panm  = (half_t*)(ws + 8192 + 3 * (size_t)PANEL_BYTES); // 16 x PANEL_BYTES
    size_t big0 = (8192 + (size_t)(3 + BATCH) * PANEL_BYTES + 255) & ~(size_t)255;
    half_t* bufA = (half_t*)(ws + big0);                            // f16 intermediates:
    half_t* bufB = (half_t*)(ws + big0 + TENSOR_HALF_BYTES);        // both fit in 192MB L2

    // Weight packing + modulation prologue (tiny vs. the convs)
    pack_static_weights<<<dim3(288, 3), 128, 0, stream>>>(conv1_w, conv2_w, convl_w,
                                                          pan1, pan2, panl);
    style_kernel<<<dim3(4), 256, 0, stream>>>(embv, mod_w, mod_b, style);
    demod_kernel<<<dim3(4), 256, 0, stream>>>(base_w, style, demod);
    pack_mod_weights<<<dim3(288, BATCH), 128, 0, stream>>>(base_w, style, demod, panm);

    const dim3 cgrid(WW / 16, HH / ROWS, BATCH);  // 12 x 48 x 16 blocks, 4 waves each

    // bufA = relu(conv1(x) + b1)                       f32 in -> f16 out
    conv3x3_wmma<float, half_t><<<cgrid, 128, 0, stream>>>(
        x, pan1, 0, conv1_b, nullptr, nullptr, nullptr, 1, bufA);
    // bufB = x + conv2(bufA) + b2                      f16 in -> f16 out
    conv3x3_wmma<half_t, half_t><<<cgrid, 128, 0, stream>>>(
        bufA, pan2, 0, conv2_b, x, nullptr, nullptr, 0, bufB);
    // bufA = modconv_b(bufB) + wn * noise              per-batch panels
    conv3x3_wmma<half_t, half_t><<<cgrid, 128, 0, stream>>>(
        bufB, panm, PANEL_ELEMS, nullptr, nullptr, noise, wnoise, 0, bufA);
    // out = leaky_relu(conv_last(bufA) + bl, 0.1)      f16 in -> f32 out
    conv3x3_wmma<half_t, float><<<cgrid, 128, 0, stream>>>(
        bufA, panl, 0, convl_b, nullptr, nullptr, nullptr, 2, out);
}
